// LSTMBlock_45140106281360
// MI455X (gfx1250) — compile-verified, matrix operand order checked by reading, not run on hardware
//
#include <hip/hip_runtime.h>
#include <hip/hip_bf16.h>
#include <math.h>
#include <stdint.h>

// ---------------------------------------------------------------------------
// LSTM fused block for MI455X (gfx1250, wave32, WMMA bf16).
// B=16384, I=H=1024.  out = [h (B*H) | c (B*H)] fp32.
// Block tile: 128 rows x 32 cols x 4 gates. 8 wave32 per block; wave w owns
// rows b0+16w..+15 and all 32 cols x 4 gates (8 accumulator tiles).
// Weights double-buffered through LDS (ds_store_b128 / ds_load_b128).
// ---------------------------------------------------------------------------

#define B_SZ  16384
#define HDIM  1024
#define IDIM  1024
#define KSTEP 32
#define LDW   40          // padded LDS row stride (elements): 32 data + 8 pad
#define NROWS 128         // LDS weight rows per buffer: 4 gates x 32 cols

typedef __bf16 bf16;
typedef __attribute__((ext_vector_type(16))) __bf16 v16bf;
typedef __attribute__((ext_vector_type(8)))  __bf16 v8bf;
typedef __attribute__((ext_vector_type(4)))  __bf16 v4bf;
typedef __attribute__((ext_vector_type(8)))  float  v8f;
typedef __attribute__((ext_vector_type(4)))  float  v4f;

union FragU { v16bf f; v8bf h[2]; };

// ---------------------------------------------------------------------------
// fp32 -> bf16 conversion (vectorized x4, NT loads); n4 = n/4.
// ---------------------------------------------------------------------------
__global__ __launch_bounds__(256) void cvt_bf16(const float* __restrict__ in,
                                                bf16* __restrict__ out, int n4) {
    int i = blockIdx.x * 256 + threadIdx.x;
    if (i < n4) {
        v4f v = __builtin_nontemporal_load((const v4f*)in + i);
        v4bf r;
        r[0] = (bf16)v[0]; r[1] = (bf16)v[1]; r[2] = (bf16)v[2]; r[3] = (bf16)v[3];
        ((v4bf*)out)[i] = r;
    }
}

// ---------------------------------------------------------------------------
// Fused 4-gate GEMM (x@Wx^T + h@Rh^T + bias) + LSTM elementwise epilogue.
// Grid: (HDIM/32, B_SZ/128), block: 256 threads (8 wave32).
// ---------------------------------------------------------------------------
__global__ __launch_bounds__(256) void lstm_wmma(
    const bf16*  __restrict__ xb,      // [B, I]  bf16
    const bf16*  __restrict__ hb,      // [B, H]  bf16
    const bf16*  __restrict__ wxb,     // [4*H, I] bf16 (flattened Wx)
    const bf16*  __restrict__ rhb,     // [4*H, H] bf16 (flattened Rh)
    const float* __restrict__ bx,      // [4, H]
    const float* __restrict__ bh,      // [4, H]
    const float* __restrict__ c_prev,  // [B, H]
    float* __restrict__ out_h,         // [B, H]
    float* __restrict__ out_c)         // [B, H]
{
    __shared__ __align__(16) bf16 lds_w[2][NROWS * LDW];   // dbl-buffered weights

    const int h0   = blockIdx.x * 32;
    const int b0   = blockIdx.y * 128;
    const int tid  = threadIdx.x;
    const int wave = tid >> 5;
    const int lane = tid & 31;

    // WMMA fragment addressing (same pattern for A and B=W^T):
    // lane L covers row (L&15), 16B chunks at k + (L>>4)*8 and +16.
    const int fr = lane & 15;
    const int fk = (lane >> 4) * 8;

    // Cooperative weight staging: 4 gates x 32 rows x 32 K bf16 = 8KB / step.
    // 256 threads: thread t loads 16 elements (two b128 chunks) of row t>>1.
    const int grow = tid >> 1;          // 0..127 : gate*32 + row
    const int sg   = grow >> 5;         // gate
    const int sr   = grow & 31;         // row within 32-col tile
    const int sc   = (tid & 1) * 16;    // element offset (0 or 16)

    v8f acc[8];
    #pragma unroll
    for (int a = 0; a < 8; ++a) acc[a] = v8f{0.f};

    for (int phase = 0; phase < 2; ++phase) {
        const bf16* A = phase ? hb  : xb;
        const bf16* W = phase ? rhb : wxb;

        const bf16* wrow = W + (size_t)(sg * HDIM + h0 + sr) * 1024 + sc;
        const bf16* arow = A + (size_t)(b0 + wave * 16 + fr) * 1024 + fk;

        const int STEPS = 1024 / KSTEP;   // 32

        // preload + stage tile 0
        v8bf w0 = *(const v8bf*)(wrow);
        v8bf w1 = *(const v8bf*)(wrow + 8);
        *(v8bf*)&lds_w[0][grow * LDW + sc]     = w0;
        *(v8bf*)&lds_w[0][grow * LDW + sc + 8] = w1;
        __syncthreads();

        for (int k = 0; k < STEPS; ++k) {
            const int cur = k & 1;
            if (k + 1 < STEPS) {
                w0 = *(const v8bf*)(wrow + (size_t)(k + 1) * KSTEP);
                w1 = *(const v8bf*)(wrow + (size_t)(k + 1) * KSTEP + 8);
            }

            // A fragment straight from global (K-contiguous rows); reused 8x.
            FragU a;
            a.h[0] = *(const v8bf*)(arow + (size_t)k * KSTEP);
            a.h[1] = *(const v8bf*)(arow + (size_t)k * KSTEP + 16);

            #pragma unroll
            for (int g = 0; g < 4; ++g) {
                #pragma unroll
                for (int c = 0; c < 2; ++c) {
                    FragU b;
                    const bf16* lw =
                        &lds_w[cur][(g * 32 + c * 16 + fr) * LDW + fk];
                    b.h[0] = *(const v8bf*)(lw);
                    b.h[1] = *(const v8bf*)(lw + 16);
                    acc[g * 2 + c] = __builtin_amdgcn_wmma_f32_16x16x32_bf16(
                        false, a.f, false, b.f,
                        (short)0, acc[g * 2 + c], false, false);
                }
            }

            if (k + 1 < STEPS) {
                *(v8bf*)&lds_w[(k + 1) & 1][grow * LDW + sc]     = w0;
                *(v8bf*)&lds_w[(k + 1) & 1][grow * LDW + sc + 8] = w1;
            }
            __syncthreads();
        }
    }

    // ------------------------- epilogue -----------------------------------
    // C/D layout: lanes 0-15 -> M=j, lanes 16-31 -> M=8+j, N = lane&15.
    const int rbase = b0 + wave * 16 + (lane >> 4) * 8;

    #pragma unroll
    for (int c = 0; c < 2; ++c) {
        const int col = h0 + c * 16 + (lane & 15);

        float bb[4];
        #pragma unroll
        for (int g = 0; g < 4; ++g)
            bb[g] = bx[g * HDIM + col] + bh[g * HDIM + col];

        #pragma unroll
        for (int j = 0; j < 8; ++j) {
            const int row = rbase + j;
            const float pz = acc[0 * 2 + c][j] + bb[0];
            const float pi = acc[1 * 2 + c][j] + bb[1];
            const float pf = acc[2 * 2 + c][j] + bb[2];
            const float po = acc[3 * 2 + c][j] + bb[3];

            const float z  = tanhf(pz);
            const float ig = 1.f / (1.f + __expf(-pi));
            const float fg = 1.f / (1.f + __expf(-pf));
            const float og = 1.f / (1.f + __expf(-po));

            const size_t idx = (size_t)row * HDIM + col;
            // Once-touched traffic: stream NT so x/h/weights stay L2-resident.
            const float cp = __builtin_nontemporal_load(c_prev + idx);
            const float cc = fg * cp + ig * z;
            __builtin_nontemporal_store(og * tanhf(cc), out_h + idx);
            __builtin_nontemporal_store(cc,             out_c + idx);
        }
    }
}

// ---------------------------------------------------------------------------
extern "C" void kernel_launch(void* const* d_in, const int* in_sizes, int n_in,
                              void* d_out, int out_size, void* d_ws, size_t ws_size,
                              hipStream_t stream) {
    (void)in_sizes; (void)n_in; (void)out_size; (void)ws_size;

    const float* x      = (const float*)d_in[0];
    const float* h_prev = (const float*)d_in[1];
    const float* c_prev = (const float*)d_in[2];
    const float* Wx     = (const float*)d_in[3];
    const float* bx     = (const float*)d_in[4];
    const float* Rh     = (const float*)d_in[5];
    const float* bh     = (const float*)d_in[6];

    float* out_h = (float*)d_out;
    float* out_c = out_h + (size_t)B_SZ * HDIM;

    // Workspace layout (bf16): x | h_prev | Wx | Rh  (~80 MB total)
    bf16* xb  = (bf16*)d_ws;
    bf16* hb  = xb  + (size_t)B_SZ * IDIM;
    bf16* wxb = hb  + (size_t)B_SZ * HDIM;
    bf16* rhb = wxb + (size_t)4 * HDIM * IDIM;

    auto cvt = [&](const float* in, bf16* out, size_t n) {
        int n4 = (int)(n / 4);
        cvt_bf16<<<(n4 + 255) / 256, 256, 0, stream>>>(in, out, n4);
    };
    cvt(x,      xb,  (size_t)B_SZ * IDIM);
    cvt(h_prev, hb,  (size_t)B_SZ * HDIM);
    cvt(Wx,     wxb, (size_t)4 * HDIM * IDIM);
    cvt(Rh,     rhb, (size_t)4 * HDIM * HDIM);

    dim3 grid(HDIM / 32, B_SZ / 128);
    lstm_wmma<<<grid, 256, 0, stream>>>(xb, hb, wxb, rhb, bx, bh, c_prev,
                                        out_h, out_c);
}
